// GraphVAE_90108413870810
// MI455X (gfx1250) — matrix operand order repaired and checked by reading, not compiled, not run on hardware
//
#include <hip/hip_runtime.h>

typedef __attribute__((ext_vector_type(16))) __bf16 v16bf;
typedef __attribute__((ext_vector_type(8)))  __bf16 v8bf;
typedef __attribute__((ext_vector_type(8)))  float  v8f;

#define NN      64
#define PITCH   68      // f32 LDS row pitch: 272B rows, 16B-aligned for float4 loads
#define WTP     72      // bf16 LDS row pitch for W^T: 144B rows, 16B-aligned
#define ITERS   20
#define NEGV    1000000.0f

__global__ __launch_bounds__(256) void graphvae_mpm_kernel(
    const float* __restrict__ A_gt,
    const float* __restrict__ vec,
    const int*   __restrict__ Rp,
    float*       __restrict__ out)
{
    __shared__ __align__(16) float  AgX[NN * PITCH];  // A (precompute), then X
    __shared__ __align__(16) float  BQ[NN * PITCH];   // B (precompute), then Q
    __shared__ __align__(16) float  NB[NN * PITCH];   // node block
    __shared__ __align__(16) __bf16 WT[NN * WTP];     // W^T (a-major) in bf16 for WMMA B
    __shared__ float  Gg[NN], Tp[NN], degA[NN], degB[NN], dBl[NN];
    __shared__ int    flags[NN];
    __shared__ float  red[8];
    __shared__ int    r_sh;
    __shared__ float  SumG, SumGtail, SumV0;

    const int tid  = threadIdx.x;
    const int lane = tid & 31;
    const int wave = tid >> 5;
    const int g    = lane >> 4;   // lane group within wave32
    const int m    = lane & 15;
    int R = Rp[0];
    if (R > NN) R = NN;

    // ---------------- Precompute: load A, build B = sigmoid(vec_to_adj) ----------------
    for (int idx = tid; idx < NN * NN; idx += 256) {
        int i = idx >> 6, j = idx & 63;
        AgX[i * PITCH + j] = A_gt[idx];
        float l;
        if (i == j) {
            l = -10.0f;
        } else {
            int ii = i < j ? i : j;
            int jj = i < j ? j : i;
            int off = ii * (NN - 1) - (ii * (ii - 1)) / 2 + (jj - ii - 1);
            l = vec[off];
        }
        BQ[i * PITCH + j] = 1.0f / (1.0f + __expf(-l));
    }
    __syncthreads();

    // r = count of rows of original A with nonzero degree
    if (tid < NN) {
        float s = 0.0f;
        for (int b = 0; b < NN; ++b) s += AgX[tid * PITCH + b];
        flags[tid] = (s > 0.0f) ? 1 : 0;
    }
    __syncthreads();
    if (tid == 0) { int rr = 0; for (int k = 0; k < NN; ++k) rr += flags[k]; r_sh = rr; }
    __syncthreads();
    const int r = r_sh;

    // diagonal override for "real" nodes
    if (tid < NN && tid < r) {
        AgX[tid * PITCH + tid] = 1.0f;
        BQ[tid * PITCH + tid]  = 1.0f;
    }
    __syncthreads();

    if (tid < NN) {
        float sA = 0.0f, sB = 0.0f;
        for (int b = 0; b < NN; ++b) {
            sA += AgX[tid * PITCH + b];
            sB += BQ[tid * PITCH + b];
        }
        degA[tid] = sA;
        degB[tid] = sB;
        dBl[tid]  = BQ[tid * PITCH + tid];
    }
    __syncthreads();

    // NB[i,a] = dA[i]*dB[a]/(|degA[i]-degB[a]|+1);  Q[a,b] = B[a,b]*dB[a]*dB[b] (in place)
    for (int idx = tid; idx < NN * NN; idx += 256) {
        int i = idx >> 6, a = idx & 63;
        float da = (i < r) ? 1.0f : 0.0f;
        NB[i * PITCH + a] = da * dBl[a] / (fabsf(degA[i] - degB[a]) + 1.0f);
        BQ[i * PITCH + a] = BQ[i * PITCH + a] * dBl[i] * dBl[a];
    }
    __syncthreads();

    // ---------------- Preload constant P fragments (WMMA A-operand, bf16) ----------------
    const int ti  = wave >> 1;          // output tile row (4x4 tile grid)
    const int ct0 = (wave & 1) * 2;     // two tile cols per wave
    const int ct1 = ct0 + 1;
    v16bf aF[2];
    for (int s = 0; s < 2; ++s) {
#pragma unroll
        for (int e = 0; e < 16; ++e) {
            int K = 16 * g + e;   // contiguous k order, matches B fragment
            int j = 32 * s + K;
            int i = 16 * ti + m;
            float p = (i < r && j < r) ? AgX[i * PITCH + j] : 0.0f;  // P in {0,1}
            aF[s][e] = (__bf16)p;
        }
    }
    __syncthreads();

    // X aliases AgX; X0 = 1/N
    float* X = AgX;
    for (int idx = tid; idx < NN * NN; idx += 256)
        X[(idx >> 6) * PITCH + (idx & 63)] = 1.0f / 64.0f;
    __syncthreads();

    // ---------------- 20 MPM iterations, fully resident in LDS ----------------
    for (int it = 0; it < ITERS; ++it) {
        // Phase 1: G[j] = -1e6*min_b X[j,b];  T+[j] = max(-1e6*min_{b>=R} X[j,b], 0)
        if (tid < NN) {
            const float4* Xr = (const float4*)&X[tid * PITCH];
            float mn_all = INFINITY;
#pragma unroll
            for (int q = 0; q < NN / 4; ++q) {
                float4 xv = Xr[q];
                mn_all = fminf(mn_all, fminf(fminf(xv.x, xv.y), fminf(xv.z, xv.w)));
            }
            float mn_tail = INFINITY;
            for (int b = R; b < NN; ++b) mn_tail = fminf(mn_tail, X[tid * PITCH + b]);
            Gg[tid] = -NEGV * mn_all;
            Tp[tid] = fmaxf(-NEGV * mn_tail, 0.0f);
        }
        __syncthreads();
        if (tid == 0) {
            float sg = 0.0f, sgt = 0.0f, sv = 0.0f;
            for (int j2 = 0; j2 < NN; ++j2) {
                sg += Gg[j2];
                if (j2 >= R) sgt += Gg[j2];
                if (j2 <  R) sv  += Tp[j2];
            }
            SumG = sg; SumGtail = sgt; SumV0 = sv;
        }

        // Phase 2: W[j,a] = max(max_{b<R,b!=a} Q[a,b]*X[j,b], 0, T+[j]) - T+[j]
        //          stored transposed: WT[a][j]
        for (int idx = tid; idx < NN * NN; idx += 256) {
            int j = idx >> 6, a = idx & 63;
            const float4* Xr = (const float4*)&X[j * PITCH];
            const float4* Qr = (const float4*)&BQ[a * PITCH];
            float u = -INFINITY;
            int nb4 = R >> 2;
            for (int q = 0; q < nb4; ++q) {
                float4 xv = Xr[q];
                float4 qv = Qr[q];
                int b0 = q << 2;
                u = fmaxf(u, (b0 + 0 == a) ? -INFINITY : qv.x * xv.x);
                u = fmaxf(u, (b0 + 1 == a) ? -INFINITY : qv.y * xv.y);
                u = fmaxf(u, (b0 + 2 == a) ? -INFINITY : qv.z * xv.z);
                u = fmaxf(u, (b0 + 3 == a) ? -INFINITY : qv.w * xv.w);
            }
            for (int b = nb4 << 2; b < R; ++b)
                u = fmaxf(u, (b == a) ? -INFINITY : BQ[a * PITCH + b] * X[j * PITCH + b]);
            float up = fmaxf(u, 0.0f);
            float w  = fmaxf(up, Tp[j]) - Tp[j];
            WT[a * WTP + j] = (__bf16)w;
        }
        __syncthreads();

        // Phase 3: E1 = P @ W via v_wmma_f32_16x16x32_bf16; B-frags = 2x ds_load_b128 each
        v8f acc0 = {0.f,0.f,0.f,0.f,0.f,0.f,0.f,0.f};
        v8f acc1 = {0.f,0.f,0.f,0.f,0.f,0.f,0.f,0.f};
#pragma unroll
        for (int s = 0; s < 2; ++s) {
            const __bf16* base0 = &WT[(16 * ct0 + m) * WTP + 32 * s + 16 * g];
            const __bf16* base1 = &WT[(16 * ct1 + m) * WTP + 32 * s + 16 * g];
            v8bf lo0 = *(const v8bf*)base0;
            v8bf hi0 = *(const v8bf*)(base0 + 8);
            v8bf lo1 = *(const v8bf*)base1;
            v8bf hi1 = *(const v8bf*)(base1 + 8);
            v16bf b0 = __builtin_shufflevector(lo0, hi0, 0,1,2,3,4,5,6,7,8,9,10,11,12,13,14,15);
            v16bf b1 = __builtin_shufflevector(lo1, hi1, 0,1,2,3,4,5,6,7,8,9,10,11,12,13,14,15);
            acc0 = __builtin_amdgcn_wmma_f32_16x16x32_bf16(
                       false, aF[s], false, b0, (short)0, acc0, false, false);
            acc1 = __builtin_amdgcn_wmma_f32_16x16x32_bf16(
                       false, aF[s], false, b1, (short)0, acc1, false, false);
        }

        // Branchless epilogue: assemble Xn, accumulate ||Xn||^2
        float sumsq = 0.0f;
#pragma unroll
        for (int t = 0; t < 2; ++t) {
            const int ct = t ? ct1 : ct0;
            v8f acc = t ? acc1 : acc0;
#pragma unroll
            for (int v = 0; v < 8; ++v) {
                int i = 16 * ti + 8 * g + v;          // f32 C/D layout: row = v + 8g
                int a = 16 * ct + m;
                float x  = X[i * PITCH + a];
                float wd = (float)WT[a * WTP + i];    // W[i,a]
                float xn_main = fmaf(x, NB[i * PITCH + a], acc[v])
                              - ((i < r) ? wd : 0.0f)
                              + (SumV0 - Tp[i]) + SumGtail;
                float xn_mask = fmaf(-NEGV, x, SumG - Gg[i]);
                float xn = (i < R && a < R) ? xn_main : xn_mask;
                sumsq += xn * xn;
                X[i * PITCH + a] = xn;
            }
        }

        // Frobenius norm: deterministic wave shuffle reduction + fixed-order partial sum
        float ws = sumsq;
#pragma unroll
        for (int o = 16; o > 0; o >>= 1) ws += __shfl_xor(ws, o, 32);
        if (lane == 0) red[wave] = ws;
        __syncthreads();
        float tot = ((red[0] + red[1]) + (red[2] + red[3]))
                  + ((red[4] + red[5]) + (red[6] + red[7]));
        float inv = 1.0f / sqrtf(tot);
        for (int idx = tid; idx < NN * NN; idx += 256)
            X[(idx >> 6) * PITCH + (idx & 63)] *= inv;
        __syncthreads();
    }

    // ---------------- write result ----------------
    for (int idx = tid; idx < NN * NN; idx += 256)
        out[idx] = X[(idx >> 6) * PITCH + (idx & 63)];
}

extern "C" void kernel_launch(void* const* d_in, const int* in_sizes, int n_in,
                              void* d_out, int out_size, void* d_ws, size_t ws_size,
                              hipStream_t stream) {
    const float* A_gt = (const float*)d_in[0];
    const float* vec  = (const float*)d_in[1];
    const int*   Rp   = (const int*)d_in[2];
    float* out = (float*)d_out;
    (void)in_sizes; (void)n_in; (void)out_size; (void)d_ws; (void)ws_size;
    graphvae_mpm_kernel<<<1, 256, 0, stream>>>(A_gt, vec, Rp, out);
}
